// GraphAttentionLayer_21406117004236
// MI455X (gfx1250) — matrix operand order repaired and checked by reading, not compiled, not run on hardware
//
#include <hip/hip_runtime.h>

// ---------------------------------------------------------------------------
// GAT layer, MI455X (gfx1250, wave32).
//   h  = x @ W                       (fp32 WMMA 16x16x4)
//   fs = h @ a[:256], fd = h @ a[256:]
//   e  = leaky_relu(fs_i + fd_j, 0.2), masked by adj, row softmax
//   h' = softmax(e) @ h              (bf16 WMMA 16x16x32, fp32 accum;
//                                     P tiles rebuilt on the fly in LDS,
//                                     the 8192x8192 matrix is never stored)
// HBM roofline: 2 passes over adj (536 MB) dominate; h kept bf16-transposed
// (4 MB) so it is L2-resident for the whole P@h GEMM.
// ---------------------------------------------------------------------------

typedef float  v2f   __attribute__((ext_vector_type(2)));
typedef float  v4f   __attribute__((ext_vector_type(4)));
typedef float  v8f   __attribute__((ext_vector_type(8)));
typedef __bf16 v8bf  __attribute__((ext_vector_type(8)));
typedef __bf16 v16bf __attribute__((ext_vector_type(16)));
typedef int    s32x2 __attribute__((ext_vector_type(2)));
typedef int    s32x4 __attribute__((ext_vector_type(4)));
typedef unsigned int u32x4 __attribute__((ext_vector_type(4)));

#define N_NODES 8192
#define F_IN    512
#define F_OUT   256

union Frag16 { u32x4 q[2]; v16bf v; };   // 16 x bf16 = 32 B
union Pack8  { u32x4 q;    v8bf  v; };   // 8 x bf16  = 16 B

// ---------------------------------------------------------------------------
// Kernel 1: h = x @ W  (fp32 WMMA). Emits fp32 h (row-major, for fs/fd) and
// bf16 h TRANSPOSED (hBt[F_OUT][N]) so pass-B B-fragments are contiguous.
// grid 1024 x 256. Each wave owns one 16x16 tile, K swept 4 at a time.
// ---------------------------------------------------------------------------
__global__ __launch_bounds__(256) void gat_gemm1(const float* __restrict__ x,
                                                 const float* __restrict__ W,
                                                 float* __restrict__ h,
                                                 __bf16* __restrict__ hBt) {
  const int t    = threadIdx.x;
  const int lane = t & 31;
  const int wv   = t >> 5;
  const int mt   = blockIdx.x >> 1;
  const int nb   = ((blockIdx.x & 1) * 8 + wv) * 16;
  const int l16  = lane & 15;
  const int hi   = lane >> 4;
  const int row  = mt * 16 + l16;   // A: all lanes hold M=0..15
  const int col  = nb + l16;        // B/C/D: N = lane % 16
  const int kk   = hi * 2;          // lanes>=16 hold K=2,3

  const float* xr = x + (size_t)row * F_IN;
  v8f acc = {};
#pragma unroll 8
  for (int kb = 0; kb < F_IN; kb += 4) {
    const int k0 = kb + kk;
    v2f a = *(const v2f*)(xr + k0);
    v2f b;
    b[0] = W[(size_t)k0 * F_OUT + col];
    b[1] = W[(size_t)(k0 + 1) * F_OUT + col];
    acc = __builtin_amdgcn_wmma_f32_16x16x4_f32(false, a, false, b,
                                                (short)0, acc, false, false);
  }
  const int mb = hi * 8;            // C/D: M = r + 8*(lane>=16)
  Pack8 pk;
#pragma unroll
  for (int r = 0; r < 8; ++r) {
    const int rr  = mt * 16 + mb + r;
    const float v = acc[r];
    h[(size_t)rr * F_OUT + col] = v;
    pk.v[r] = (__bf16)v;
  }
  // transposed bf16 store: 8 consecutive k-rows at this column, one b128
  *(u32x4*)(hBt + (size_t)col * N_NODES + mt * 16 + mb) = pk.q;
}

// ---------------------------------------------------------------------------
// Kernel 2: fs = h @ a_l, fd = h @ a_r  (one thread per node row, b128 loads)
// ---------------------------------------------------------------------------
__global__ __launch_bounds__(256) void gat_fsfd(const float* __restrict__ h,
                                                const float* __restrict__ a,
                                                float* __restrict__ fs,
                                                float* __restrict__ fd) {
  const int i = blockIdx.x * 256 + threadIdx.x;
  const v4f* hr = (const v4f*)(h + (size_t)i * F_OUT);
  const v4f* al = (const v4f*)(a);
  const v4f* ar = (const v4f*)(a + F_OUT);
  float s = 0.f, d = 0.f;
#pragma unroll 4
  for (int k = 0; k < F_OUT / 4; ++k) {
    const v4f v = hr[k], bl = al[k], br = ar[k];
#pragma unroll
    for (int e = 0; e < 4; ++e) {
      s = fmaf(v[e], bl[e], s);
      d = fmaf(v[e], br[e], d);
    }
  }
  fs[i] = s;
  fd[i] = d;
}

// ---------------------------------------------------------------------------
// Kernel 3: per-row masked online max + softmax denominator.
// One block per row; adj streamed as b128 (268 MB total, HBM-bound).
// ---------------------------------------------------------------------------
__global__ __launch_bounds__(256) void gat_rowstats(const int* __restrict__ adj,
                                                    const float* __restrict__ fs,
                                                    const float* __restrict__ fd,
                                                    float* __restrict__ mrow,
                                                    float* __restrict__ invs) {
  __shared__ float sm[256], ss[256];
  const int i = blockIdx.x;
  const int t = threadIdx.x;
  const float fsi = fs[i];
  const int* ar = adj + (size_t)i * N_NODES;

  float m = -__builtin_inff(), s = 0.f;
  for (int j = t * 4; j < N_NODES; j += 1024) {
    const s32x4 av  = *(const s32x4*)(ar + j);
    const v4f   fdv = *(const v4f*)(fd + j);
#pragma unroll
    for (int e = 0; e < 4; ++e) {
      if (av[e] > 0) {
        float z = fsi + fdv[e];
        z = z > 0.f ? z : 0.2f * z;               // leaky_relu(0.2)
        if (z > m) { s *= __expf(m - z); m = z; } // exp(-inf)=0 on first hit
        s += __expf(z - m);
      }
    }
  }
  sm[t] = m; ss[t] = s;
  __syncthreads();
  for (int off = 128; off > 0; off >>= 1) {
    if (t < off) {
      const float m1 = sm[t], s1 = ss[t];
      const float m2 = sm[t + off], s2 = ss[t + off];
      const float mx = fmaxf(m1, m2);
      float sn = 0.f;
      if (mx > -__builtin_inff())
        sn = s1 * __expf(m1 - mx) + s2 * __expf(m2 - mx);
      sm[t] = mx; ss[t] = sn;
    }
    __syncthreads();
  }
  if (t == 0) {
    mrow[i] = sm[0];
    invs[i] = ss[0] > 0.f ? 1.0f / ss[0] : 0.f;
  }
}

// ---------------------------------------------------------------------------
// Kernel 4: h' = P @ h. grid 512 (16-row blocks) x 256 threads (8 waves).
// Ping-pong 16x32 bf16 P tile in LDS -> single barrier per k-step.
// Wave w owns output cols [32w,32w+32): 2 bf16 WMMAs/step, 256 steps.
// B fragments are contiguous 32 B reads from transposed bf16 h.
// ---------------------------------------------------------------------------
__global__ __launch_bounds__(256) void gat_gemm2(const int* __restrict__ adj,
                                                 const float* __restrict__ fs,
                                                 const float* __restrict__ fd,
                                                 const float* __restrict__ mrow,
                                                 const float* __restrict__ invs,
                                                 const __bf16* __restrict__ hBt,
                                                 float* __restrict__ out) {
  __shared__ __align__(64) __bf16 Pt[2][16 * 32];
  __shared__ float fsS[16], mS[16], iS[16];
  const int t  = threadIdx.x;
  const int ib = blockIdx.x * 16;
  if (t < 16) {
    fsS[t] = fs[ib + t];
    mS[t]  = mrow[ib + t];
    iS[t]  = invs[ib + t];
  }
  __syncthreads();

  const int lane  = t & 31;
  const int wv    = t >> 5;
  const int l16   = lane & 15;
  const int hi    = lane >> 4;
  const int prow  = t >> 4;          // P-build: row 0..15
  const int pcol  = (t & 15) * 2;    // P-build: cols 2t, 2t+1
  const int col0  = wv * 32 + l16;   // B/C/D: N = lane % 16
  const int col1  = col0 + 16;
  const int khalf = hi * 16;         // B: lanes>=16 hold K=16..31

  const int* aprow = adj + (size_t)(ib + prow) * N_NODES + pcol;
  const float fsr = fsS[prow], mm = mS[prow], is = iS[prow];
  const __bf16* hb0 = hBt + (size_t)col0 * N_NODES + khalf;
  const __bf16* hb1 = hBt + (size_t)col1 * N_NODES + khalf;

  v8f c0 = {}, c1 = {};

  auto step = [&](int kb, int buf) {
    // ---- build 16x32 probability tile in LDS (2 elements / thread) ----
    const s32x2 aa  = *(const s32x2*)(aprow + kb);
    const v2f   fdv = *(const v2f*)(fd + kb + pcol);
    float z0 = fsr + fdv[0];
    float z1 = fsr + fdv[1];
    z0 = z0 > 0.f ? z0 : 0.2f * z0;
    z1 = z1 > 0.f ? z1 : 0.2f * z1;
    const float p0 = (aa[0] > 0) ? __expf(z0 - mm) * is : 0.f;
    const float p1 = (aa[1] > 0) ? __expf(z1 - mm) * is : 0.f;
    __bf16* P = Pt[buf];
    P[prow * 32 + pcol]     = (__bf16)p0;
    P[prow * 32 + pcol + 1] = (__bf16)p1;
    __builtin_prefetch(aprow + kb + 64, 0, 1);   // next adj tile -> L2/WGP
    __syncthreads();

    // ---- A fragment: row l16, K contiguous hi*16..hi*16+15 ----
    Frag16 af;
    const u32x4* rp = (const u32x4*)(P + l16 * 32);
    af.q[0] = rp[2 * hi];
    af.q[1] = rp[2 * hi + 1];

    // ---- B fragments: contiguous 32 B from transposed bf16 h ----
    Frag16 b0, b1;
    const u32x4* bp0 = (const u32x4*)(hb0 + kb);
    const u32x4* bp1 = (const u32x4*)(hb1 + kb);
    b0.q[0] = bp0[0]; b0.q[1] = bp0[1];
    b1.q[0] = bp1[0]; b1.q[1] = bp1[1];

    c0 = __builtin_amdgcn_wmma_f32_16x16x32_bf16(false, af.v, false, b0.v,
                                                 (short)0, c0, false, false);
    c1 = __builtin_amdgcn_wmma_f32_16x16x32_bf16(false, af.v, false, b1.v,
                                                 (short)0, c1, false, false);
    // no trailing barrier: next step writes the other LDS buffer; our ds
    // reads are consumed by the WMMAs before this thread's next write, and
    // the next step's barrier orders everyone before buffer reuse.
  };

  for (int kb = 0; kb < N_NODES; kb += 64) {
    step(kb, 0);
    step(kb + 32, 1);
  }

  const int mb = hi * 8;             // C/D: M = r + 8*(lane>=16)
#pragma unroll
  for (int r = 0; r < 8; ++r) {
    const int row = ib + mb + r;
    out[(size_t)row * F_OUT + col0]      = c0[r];
    out[(size_t)row * F_OUT + col0 + 16] = c1[r];
  }
}

// ---------------------------------------------------------------------------
extern "C" void kernel_launch(void* const* d_in, const int* in_sizes, int n_in,
                              void* d_out, int out_size, void* d_ws, size_t ws_size,
                              hipStream_t stream) {
  (void)in_sizes; (void)n_in; (void)out_size; (void)ws_size;
  const float* x   = (const float*)d_in[0];
  const int*   adj = (const int*)d_in[1];
  const float* W   = (const float*)d_in[2];
  const float* a   = (const float*)d_in[3];
  float* out = (float*)d_out;

  // workspace layout (~12.7 MB)
  char* ws = (char*)d_ws;
  float*  h    = (float*)(ws);                                   // 8 MB fp32 h
  __bf16* hBt  = (__bf16*)(ws + (size_t)N_NODES * F_OUT * 4);    // 4 MB bf16 h^T
  float*  fs   = (float*)(ws + (size_t)N_NODES * F_OUT * 6);
  float*  fd   = fs + N_NODES;
  float*  mrow = fd + N_NODES;
  float*  invs = mrow + N_NODES;

  gat_gemm1<<<dim3(1024), dim3(256), 0, stream>>>(x, W, h, hBt);
  gat_fsfd<<<dim3(N_NODES / 256), dim3(256), 0, stream>>>(h, a, fs, fd);
  gat_rowstats<<<dim3(N_NODES), dim3(256), 0, stream>>>(adj, fs, fd, mrow, invs);
  gat_gemm2<<<dim3(N_NODES / 16), dim3(256), 0, stream>>>(adj, fs, fd, mrow,
                                                          invs, hBt, out);
}
